// V8Model_21449066676284
// MI455X (gfx1250) — matrix operand order repaired and checked by reading, not compiled, not run on hardware
//
#include <hip/hip_runtime.h>
#include <cstdint>

// ---------------------------------------------------------------------------
// Model constants
// ---------------------------------------------------------------------------
#define BSZ      4
#define SEQ      512
#define VOCAB    32000
#define NC       16
#define DMEM     64
#define DMODEL   1024          // NC * DMEM
#define DFF      4096
#define NNEUR    32
#define MAXCONN  16
#define NMHID    256
#define ACTDIM   82            // DMEM + MAXCONN + 2
#define ROWS     (BSZ * SEQ)   // 2048 token rows

typedef __attribute__((ext_vector_type(16))) __bf16 v16bf;
typedef __attribute__((ext_vector_type(8)))  float  v8f;

union Frag16 { v16bf v; unsigned int u[8]; };

__device__ __forceinline__ unsigned short f32_to_bf16(float f) {
    unsigned int u = __float_as_uint(f);
    unsigned int r = u + 0x7FFFu + ((u >> 16) & 1u);   // round-to-nearest-even
    return (unsigned short)(r >> 16);
}

__device__ __forceinline__ float gelu_tanh(float x) {
    float x3 = x * x * x;
    return 0.5f * x * (1.0f + tanhf(0.7978845608028654f * (x + 0.044715f * x3)));
}

// ---------------------------------------------------------------------------
// f32 [K][N] -> bf16 [N][K]  (transposed so B fragments load K-contiguous)
// ---------------------------------------------------------------------------
__global__ void transpose_to_bf16(const float* __restrict__ src,
                                  unsigned short* __restrict__ dst,
                                  int K, int N) {
    size_t total = (size_t)K * (size_t)N;
    for (size_t i = (size_t)blockIdx.x * blockDim.x + threadIdx.x; i < total;
         i += (size_t)gridDim.x * blockDim.x) {
        int k = (int)(i / N), n = (int)(i % N);
        dst[(size_t)n * K + k] = f32_to_bf16(src[i]);
    }
}

// ---------------------------------------------------------------------------
// Embedding gather: emb f32 + bf16 copies
// ---------------------------------------------------------------------------
__global__ void embed_kernel(const int* __restrict__ ids,
                             const float* __restrict__ E,
                             float* __restrict__ embF,
                             unsigned short* __restrict__ embBf) {
    int row = blockIdx.x;                    // 0..ROWS-1
    int id  = ids[row];
    const float* src = E + (size_t)id * DMODEL;
    for (int i = threadIdx.x; i < DMODEL; i += blockDim.x) {
        float v = src[i];
        embF [(size_t)row * DMODEL + i] = v;
        embBf[(size_t)row * DMODEL + i] = f32_to_bf16(v);
    }
}

// ---------------------------------------------------------------------------
// WMMA bf16 GEMM: C(M,N) = A(M,K) * Bt(N,K)^T, f32 accumulate.
// Each wave computes a 16x64 strip (4 N-tiles): the A fragment is loaded once
// per K-step (32 deep) and reused by 4 v_wmma_f32_16x16x32_bf16 issues.
// Requires N % 64 == 0 (true for 4096 / 1024 / 32000).
// Epilogue modes: 0 gelu->bf16, 1 +res->f32, 2 +res->bf16, 3 ->f32
// ---------------------------------------------------------------------------
__global__ void gemm_wmma(const unsigned short* __restrict__ A,
                          const unsigned short* __restrict__ Bt,
                          const float* __restrict__ res,
                          float* __restrict__ outF,
                          unsigned short* __restrict__ outBf,
                          int M, int N, int K, int mode) {
    const int lane = threadIdx.x & 31;
    const int half = lane >> 4;              // 0: lanes 0-15, 1: lanes 16-31
    const int lrow = lane & 15;
    const int wid  = (int)(blockIdx.x * (blockDim.x >> 5) + (threadIdx.x >> 5));
    const int grpN = N >> 6;                 // groups of 4 N-tiles
    const int grpTotal = (M >> 4) * grpN;
    if (wid >= grpTotal) return;
    const int tileM = wid / grpN;
    const int tileN = (wid % grpN) << 2;     // first of 4 consecutive 16-col tiles

    const unsigned short* Arow = A + (size_t)(tileM * 16 + lrow) * K;
    const unsigned short* Brow[4];
#pragma unroll
    for (int j = 0; j < 4; ++j)
        Brow[j] = Bt + (size_t)((tileN + j) * 16 + lrow) * K;   // column = lrow

    v8f acc0 = {}, acc1 = {}, acc2 = {}, acc3 = {};
    for (int k0 = 0; k0 < K; k0 += 32) {
        Frag16 af, b0, b1, b2, b3;
#pragma unroll
        for (int v = 0; v < 8; ++v) {
            // A: contiguous k order chosen here (same as B): VGPR v holds K = 2*v + 16*half (+1 in hi16)
            int ka = (v << 1) + (half << 4);
            af.u[v] = *reinterpret_cast<const unsigned int*>(Arow + k0 + ka);
            // B: same contiguous k order as A: lanes 0-15 K=0..15, lanes 16-31 K=16..31, 2 per VGPR
            int kb = (v << 1) + (half << 4);
            b0.u[v] = *reinterpret_cast<const unsigned int*>(Brow[0] + k0 + kb);
            b1.u[v] = *reinterpret_cast<const unsigned int*>(Brow[1] + k0 + kb);
            b2.u[v] = *reinterpret_cast<const unsigned int*>(Brow[2] + k0 + kb);
            b3.u[v] = *reinterpret_cast<const unsigned int*>(Brow[3] + k0 + kb);
        }
        acc0 = __builtin_amdgcn_wmma_f32_16x16x32_bf16(
                   false, af.v, false, b0.v, (short)0, acc0, false, false);
        acc1 = __builtin_amdgcn_wmma_f32_16x16x32_bf16(
                   false, af.v, false, b1.v, (short)0, acc1, false, false);
        acc2 = __builtin_amdgcn_wmma_f32_16x16x32_bf16(
                   false, af.v, false, b2.v, (short)0, acc2, false, false);
        acc3 = __builtin_amdgcn_wmma_f32_16x16x32_bf16(
                   false, af.v, false, b3.v, (short)0, acc3, false, false);
    }

#pragma unroll
    for (int j = 0; j < 4; ++j) {
        v8f acc = (j == 0) ? acc0 : (j == 1) ? acc1 : (j == 2) ? acc2 : acc3;
#pragma unroll
        for (int r = 0; r < 8; ++r) {
            int row = tileM * 16 + r + (half << 3);  // C/D: VGPR r -> M=r (+8 hi half)
            int col = (tileN + j) * 16 + lrow;
            size_t idx = (size_t)row * N + col;
            float x = acc[r];
            if (mode == 0) {
                outBf[idx] = f32_to_bf16(gelu_tanh(x));
            } else if (mode == 1) {
                outF[idx] = x + res[idx];
            } else if (mode == 2) {
                outBf[idx] = f32_to_bf16(x + res[idx]);
            } else {
                outF[idx] = x;
            }
        }
    }
}

// ---------------------------------------------------------------------------
// aux_loss = mean(h*h)
// ---------------------------------------------------------------------------
__global__ void aux_reduce(const float* __restrict__ h, float* __restrict__ aux, int n) {
    __shared__ float red[256];
    float s = 0.0f;
    for (size_t i = (size_t)blockIdx.x * blockDim.x + threadIdx.x; i < (size_t)n;
         i += (size_t)gridDim.x * blockDim.x) {
        float v = h[i];
        s += v * v;
    }
    red[threadIdx.x] = s;
    __syncthreads();
    for (int o = 128; o > 0; o >>= 1) {
        if (threadIdx.x < (unsigned)o) red[threadIdx.x] += red[threadIdx.x + o];
        __syncthreads();
    }
    if (threadIdx.x == 0)
        atomicAdd(aux, red[0] * (1.0f / (float)(ROWS * DMODEL)));
}

// ---------------------------------------------------------------------------
// Sequential memory scan: one workgroup per batch, state in LDS.
// ---------------------------------------------------------------------------
__global__ void scan_kernel(const float* __restrict__ h,
                            const float* __restrict__ emb,
                            const int* __restrict__ ids,
                            const float* __restrict__ prim_init,
                            const float* __restrict__ nm_W1,
                            const float* __restrict__ nm_b1,
                            const float* __restrict__ nm_Wa,
                            unsigned short* __restrict__ memBf) {
    const int b   = blockIdx.x;
    const int tid = threadIdx.x;

    __shared__ float sPrim[NNEUR][DMEM];       // 8 KB
    __shared__ float sThresh[NNEUR][MAXCONN];  // 2 KB
    __shared__ float sTemp[NNEUR], sDecay[NNEUR];
    __shared__ float sCC[NC][DMEM];            // 4 KB
    __shared__ float sScores[NC][NNEUR];       // 2 KB
    __shared__ float sAttnG[NC][NNEUR];        // 2 KB
    __shared__ float sMemOut[NC][DMEM];        // 4 KB
    __shared__ float sMsAccum[DMEM], sMsPrev[DMEM];
    __shared__ float sDsig[NNEUR], sThreshMean[NNEUR], sExpTemp[NNEUR], sNorm[NNEUR];
    __shared__ float sHid[NNEUR][NMHID];       // 32 KB

    for (int i = tid; i < NNEUR * DMEM; i += 256)
        sPrim[i >> 6][i & 63] = prim_init[b * NNEUR * DMEM + i];
    for (int i = tid; i < NNEUR * MAXCONN; i += 256)
        sThresh[i >> 4][i & 15] = 0.0f;
    if (tid < NNEUR) { sTemp[tid] = 0.0f; sDecay[tid] = 0.0f; }
    if (tid < DMEM)  { sMsAccum[tid] = 0.0f; sMsPrev[tid] = 0.0f; }
    __syncthreads();

    for (int t = 0; t < SEQ; ++t) {
        const size_t rowOff = ((size_t)b * SEQ + t) * DMODEL;

        // eot_prev reset (uniform branch per workgroup)
        bool ep = (t > 0) && (ids[b * SEQ + t - 1] == 0);
        if (ep) {
            for (int i = tid; i < NNEUR * DMEM; i += 256)
                sPrim[i >> 6][i & 63] = prim_init[b * NNEUR * DMEM + i];
            for (int i = tid; i < NNEUR * MAXCONN; i += 256)
                sThresh[i >> 4][i & 15] = 0.0f;
            if (tid < NNEUR) { sTemp[tid] = 0.0f; sDecay[tid] = 0.0f; }
        }

        // cc_t = h[b,t] reshaped (16,64)
        for (int i = tid; i < DMODEL; i += 256)
            sCC[i >> 6][i & 63] = h[rowOff + i];

        // window-mean bookkeeping: snapshot previous window at window start
        if ((t & 15) == 0 && tid < DMEM) {
            sMsPrev[tid]  = (t == 0) ? 0.0f : sMsAccum[tid] * (1.0f / 256.0f);
            sMsAccum[tid] = 0.0f;
        }
        __syncthreads();

        // accumulate current-step surprise (h - emb) into window accumulator
        if (tid < DMEM) {
            float s = 0.0f;
            for (int c = 0; c < NC; ++c)
                s += sCC[c][tid] - emb[rowOff + c * DMEM + tid];
            sMsAccum[tid] += s;
        }
        if (tid < NNEUR) {
            float tm = 0.0f;
            for (int j = 0; j < MAXCONN; ++j) tm += sThresh[tid][j];
            sThreshMean[tid] = tm * (1.0f / (float)MAXCONN);
            sDsig[tid]    = 1.0f / (1.0f + expf(-sDecay[tid]));
            sExpTemp[tid] = expf(sTemp[tid]);
        }
        __syncthreads();

        // scores[c,n] = cc . prim / sqrt(64)
        for (int i = tid; i < NC * NNEUR; i += 256) {
            int c = i >> 5, n = i & 31;
            float s = 0.0f;
            for (int d = 0; d < DMEM; ++d) s += sCC[c][d] * sPrim[n][d];
            sScores[c][n] = s * 0.125f;
        }
        __syncthreads();

        // softmax over n (with per-neuron temperature) * sigmoid gate
        if (tid < NC) {
            int c = tid;
            float lg[NNEUR];
            float mx = -3.4e38f;
            for (int n = 0; n < NNEUR; ++n) {
                lg[n] = sScores[c][n] * sExpTemp[n];
                mx = fmaxf(mx, lg[n]);
            }
            float sum = 0.0f;
            for (int n = 0; n < NNEUR; ++n) { lg[n] = expf(lg[n] - mx); sum += lg[n]; }
            float inv = 1.0f / sum;
            for (int n = 0; n < NNEUR; ++n) {
                float gate = 1.0f / (1.0f + expf(-(sScores[c][n] - sThreshMean[n])));
                sAttnG[c][n] = lg[n] * inv * gate;
            }
        }
        __syncthreads();

        // mem_out[c,d] = sum_n attn_g * prim   (uses pre-update prim)
        for (int i = tid; i < NC * DMEM; i += 256) {
            int c = i >> 6, d = i & 63;
            float s = 0.0f;
            for (int n = 0; n < NNEUR; ++n) s += sAttnG[c][n] * sPrim[n][d];
            sMemOut[c][d] = s;
        }
        __syncthreads();

        // prim_new = dsig*prim + (1-dsig)*writes   (in place)
        for (int i = tid; i < NNEUR * DMEM; i += 256) {
            int n = i >> 6, d = i & 63;
            float w = 0.0f;
            for (int c = 0; c < NC; ++c) w += sAttnG[c][n] * sCC[c][d];
            w *= (1.0f / (float)NC);
            float ds = sDsig[n];
            sPrim[n][d] = ds * sPrim[n][d] + (1.0f - ds) * w;
        }
        // emit mem_out as bf16 for the Wmem GEMM
        for (int i = tid; i < DMODEL; i += 256)
            memBf[rowOff + i] = f32_to_bf16(sMemOut[i >> 6][i & 63]);
        __syncthreads();

        // neuromodulator MLP only alters state on action steps (t % 16 == 0)
        if ((t & 15) == 0) {
            if (tid < NNEUR) {
                float s = 0.0f;
                for (int d = 0; d < DMEM; ++d) s += sPrim[tid][d] * sPrim[tid][d];
                sNorm[tid] = sqrtf(s);
            }
            __syncthreads();
            // hid[n,j] = tanh(obs . nm_W1 + b1), obs built on the fly
            for (int i = tid; i < NNEUR * NMHID; i += 256) {
                int n = i >> 8, j = i & 255;
                float s = nm_b1[j];
                for (int d = 0; d < DMEM; ++d) {
                    float p = sPrim[n][d], m = sMsPrev[d];
                    s += p       * nm_W1[(d)       * NMHID + j];
                    s += m       * nm_W1[(64  + d) * NMHID + j];
                    s += (p - m) * nm_W1[(128 + d) * NMHID + j];
                    s += (p * m) * nm_W1[(192 + d) * NMHID + j];
                }
                s += sTemp[n]       * nm_W1[256 * NMHID + j];
                s += sDecay[n]      * nm_W1[257 * NMHID + j];
                s += sThreshMean[n] * nm_W1[258 * NMHID + j];
                s += sNorm[n]       * nm_W1[259 * NMHID + j];
                sHid[n][j] = tanhf(s);
            }
            __syncthreads();
            // act = clip(hid @ nm_Wa, +-0.1), applied in place (unique targets)
            for (int i = tid; i < NNEUR * ACTDIM; i += 256) {
                int n = i / ACTDIM, o = i % ACTDIM;
                float s = 0.0f;
                for (int j = 0; j < NMHID; ++j) s += sHid[n][j] * nm_Wa[j * ACTDIM + o];
                s = fminf(fmaxf(s, -0.1f), 0.1f);
                if (o < DMEM)                 sPrim[n][o] += s;
                else if (o < DMEM + MAXCONN)  sThresh[n][o - DMEM] += s;
                else if (o == DMEM + MAXCONN) sTemp[n] += s;
                else                          sDecay[n] += s;
            }
        }
        __syncthreads();
    }
}

// ---------------------------------------------------------------------------
// Host launcher
// ---------------------------------------------------------------------------
extern "C" void kernel_launch(void* const* d_in, const int* in_sizes, int n_in,
                              void* d_out, int out_size, void* d_ws, size_t ws_size,
                              hipStream_t stream) {
    (void)in_sizes; (void)n_in; (void)ws_size;

    const int*   ids      = (const int*)  d_in[0];
    const float* E        = (const float*)d_in[1];
    const float* W1       = (const float*)d_in[2];
    const float* W2       = (const float*)d_in[3];
    const float* Wmem     = (const float*)d_in[4];
    const float* Wout     = (const float*)d_in[5];
    const float* nm_W1    = (const float*)d_in[6];
    const float* nm_b1    = (const float*)d_in[7];
    const float* nm_Wa    = (const float*)d_in[8];
    const float* primInit = (const float*)d_in[9];

    float* logits = (float*)d_out;                       // (BS,T,VOCAB) row-major
    float* auxPtr = (float*)d_out + (size_t)out_size - 1;

    // workspace carve-up (256B aligned)
    char* ws = (char*)d_ws;
    size_t off = 0;
    auto carve = [&](size_t bytes) {
        size_t o = off;
        off = (off + bytes + 255) & ~(size_t)255;
        return ws + o;
    };
    float*          embF   = (float*)         carve((size_t)ROWS * DMODEL * 4);
    unsigned short* embBf  = (unsigned short*)carve((size_t)ROWS * DMODEL * 2);
    unsigned short* actBf  = (unsigned short*)carve((size_t)ROWS * DFF    * 2);
    float*          hF     = (float*)         carve((size_t)ROWS * DMODEL * 4);
    unsigned short* memBf  = (unsigned short*)carve((size_t)ROWS * DMODEL * 2);
    unsigned short* hmemBf = (unsigned short*)carve((size_t)ROWS * DMODEL * 2);
    unsigned short* W1t    = (unsigned short*)carve((size_t)DMODEL * DFF    * 2);
    unsigned short* W2t    = (unsigned short*)carve((size_t)DFF    * DMODEL * 2);
    unsigned short* Wmemt  = (unsigned short*)carve((size_t)DMODEL * DMODEL * 2);
    unsigned short* Woutt  = (unsigned short*)carve((size_t)DMODEL * VOCAB  * 2);

    // 1) weight convert + transpose to bf16 [N][K]
    transpose_to_bf16<<<4096, 256, 0, stream>>>(W1,   W1t,   DMODEL, DFF);
    transpose_to_bf16<<<4096, 256, 0, stream>>>(W2,   W2t,   DFF,    DMODEL);
    transpose_to_bf16<<<1024, 256, 0, stream>>>(Wmem, Wmemt, DMODEL, DMODEL);
    transpose_to_bf16<<<8192, 256, 0, stream>>>(Wout, Woutt, DMODEL, VOCAB);

    // 2) embedding gather
    embed_kernel<<<ROWS, 256, 0, stream>>>(ids, E, embF, embBf);

    // 3) FFN: act = gelu(emb @ W1) -> bf16
    {
        int grps = (ROWS / 16) * (DFF / 64);
        gemm_wmma<<<(grps + 7) / 8, 256, 0, stream>>>(
            embBf, W1t, nullptr, nullptr, actBf, ROWS, DFF, DMODEL, 0);
    }
    // 4) h = act @ W2 + emb -> f32
    {
        int grps = (ROWS / 16) * (DMODEL / 64);
        gemm_wmma<<<(grps + 7) / 8, 256, 0, stream>>>(
            actBf, W2t, embF, hF, nullptr, ROWS, DMODEL, DFF, 1);
    }

    // 5) aux_loss = mean(h*h)
    hipMemsetAsync(auxPtr, 0, sizeof(float), stream);
    aux_reduce<<<256, 256, 0, stream>>>(hF, auxPtr, ROWS * DMODEL);

    // 6) sequential memory scan -> mem_signals (bf16)
    scan_kernel<<<BSZ, 256, 0, stream>>>(hF, embF, ids, primInit,
                                         nm_W1, nm_b1, nm_Wa, memBf);

    // 7) hmem = h + mem @ Wmem -> bf16
    {
        int grps = (ROWS / 16) * (DMODEL / 64);
        gemm_wmma<<<(grps + 7) / 8, 256, 0, stream>>>(
            memBf, Wmemt, hF, nullptr, hmemBf, ROWS, DMODEL, DMODEL, 2);
    }
    // 8) logits = hmem @ Wout -> f32 (into d_out)
    {
        int grps = (ROWS / 16) * (VOCAB / 64);
        gemm_wmma<<<(grps + 7) / 8, 256, 0, stream>>>(
            hmemBf, Woutt, nullptr, logits, nullptr, ROWS, VOCAB, DMODEL, 3);
    }
}
